// AttGNN_76141180224214
// MI455X (gfx1250) — compile-verified, matrix operand order checked by reading, not run on hardware
//
#include <hip/hip_runtime.h>
#include <hip/hip_bf16.h>

// ---------------------------------------------------------------------------
// AttGNN on MI455X (gfx1250): bf16 WMMA (16x16x32) with f32 accumulation,
// flash-style fused attention softmax so the 8192x8192 attention matrix is
// never materialized. HBM-bound on streaming `sup` (2 x 256MB); attention
// element math uses the native V_TANH_F32 / V_EXP_F32 trans ops.
// ---------------------------------------------------------------------------

#define NN   8192
#define DIN  128
#define DGAT 64
#define DGCN 64

typedef __bf16 bf16_t;
typedef __attribute__((ext_vector_type(16))) __bf16 v16bf;
typedef __attribute__((ext_vector_type(8)))  __bf16 v8bf;
typedef __attribute__((ext_vector_type(8)))  float  v8f;

static __device__ __forceinline__ v8f wmma_bf16(v16bf a, v16bf b, v8f c) {
  // D = A(16x32 bf16) * B(32x16 bf16) + C(16x16 f32)
  return __builtin_amdgcn_wmma_f32_16x16x32_bf16(false, a, false, b,
                                                 (short)0, c, false, false);
}

// Native CDNA5 transcendental tanh (V_TANH_F32), 1 trans op vs ~20 VALU ocml.
static __device__ __forceinline__ float fast_tanh(float x) {
#if __has_builtin(__builtin_amdgcn_tanhf)
  return __builtin_amdgcn_tanhf(x);
#else
  float r;
  asm("v_tanh_f32 %0, %1" : "=v"(r) : "v"(x));
  return r;
#endif
}

// A-fragment per-lane K offsets (16-bit A 16x32 layout):
//   lanes 0-15 : row = lane,     K = {0..7, 16..23}
//   lanes 16-31: row = lane-16,  K = {8..15, 24..31}
// element e of v16bf  ->  K = (e<8 ? e : e+8) + (hi ? 8 : 0)

// ---------------------------------------------------------------------------
// K1: h = feat @ W_map   (8192x128x64). Also emit hT (bf16, [64][8192]).
// One wave per 16x16 output tile. 2048 tiles, 2 waves/block.
// ---------------------------------------------------------------------------
__global__ void k_map(const float* __restrict__ feat,
                      const float* __restrict__ Wmap,
                      float* __restrict__ h, bf16_t* __restrict__ hT) {
  const int lane  = threadIdx.x & 31;
  const int wv    = threadIdx.x >> 5;
  const int tile  = blockIdx.x * 2 + wv;
  const int itile = tile >> 2, ctile = tile & 3;
  const int ibase = itile * 16, cbase = ctile * 16;
  const int hi = lane >> 4, lr = lane & 15;
  const int kb0 = hi ? 8 : 0;

  v8f acc = {};
  const float* frow = feat + (size_t)(ibase + lr) * DIN;
#pragma unroll
  for (int ch = 0; ch < 4; ++ch) {
    const int kb = ch * 32 + kb0;
    v16bf a, b;
#pragma unroll
    for (int e = 0; e < 8; ++e) {
      a[e]     = (bf16_t)frow[kb + e];
      a[8 + e] = (bf16_t)frow[kb + 16 + e];
      b[e]     = (bf16_t)Wmap[(size_t)(kb + e) * DGAT + cbase + lr];
      b[8 + e] = (bf16_t)Wmap[(size_t)(kb + 16 + e) * DGAT + cbase + lr];
    }
    acc = wmma_bf16(a, b, acc);
  }
  // C/D layout: lane lr holds col cbase+lr, rows hi*8 + m
#pragma unroll
  for (int m = 0; m < 8; ++m) {
    const int row = ibase + hi * 8 + m;
    const int col = cbase + lr;
    const float v = acc[m];
    h[(size_t)row * DGAT + col] = v;
    hT[(size_t)col * NN + row]  = (bf16_t)v;
  }
}

// ---------------------------------------------------------------------------
// K1b: s = h @ U + bias (bias folded), t = h @ V  (per-row dot products)
// ---------------------------------------------------------------------------
__global__ void k_scores(const float* __restrict__ h, const float* __restrict__ U,
                         const float* __restrict__ V, const float* __restrict__ bmap,
                         float* __restrict__ s, float* __restrict__ t) {
  const int row = blockIdx.x * blockDim.x + threadIdx.x;
  if (row >= NN) return;
  const float* hr = h + (size_t)row * DGAT;
  float as = 0.f, at = 0.f;
#pragma unroll
  for (int c = 0; c < DGAT; ++c) { const float v = hr[c]; as += v * U[c]; at += v * V[c]; }
  s[row] = as + bmap[0];   // att(i,j) = tanh(s'_j + t_i)
  t[row] = at;
}

// ---------------------------------------------------------------------------
// K2: fused masked-tanh attention + online softmax + (att @ h) -> gatT (bf16)
// One wave per 16-row tile; stream j in chunks of 32 (= WMMA K).
// ---------------------------------------------------------------------------
__global__ void k_att(const float* __restrict__ sup, const float* __restrict__ s,
                      const float* __restrict__ t,
                      const bf16_t* __restrict__ hT, bf16_t* __restrict__ gatT) {
  const int lane  = threadIdx.x & 31;
  const int wv    = threadIdx.x >> 5;
  const int itile = blockIdx.x * 2 + wv;
  const int ibase = itile * 16;
  const int hi = lane >> 4, lr = lane & 15;
  const int kb0 = hi ? 8 : 0;

  const float trow = t[ibase + lr];         // this lane owns row (ibase+lr)'s stats
  const float* srow = sup + (size_t)(ibase + lr) * NN;

  float mrun = -100.0f, lrun = 0.f;         // mask value = floor of all logits
  v8f acc[4] = {};

  for (int jb = 0; jb < NN; jb += 32) {
    __builtin_prefetch(srow + jb + 512, 0, 1);   // global_prefetch_b8 down the sup stream
    // masked attention logits for this lane's 16 (row,K) slots
    float p[16];
#pragma unroll
    for (int run = 0; run < 2; ++run) {
      const int kb = jb + kb0 + run * 16;
#pragma unroll
      for (int e = 0; e < 8; ++e) {
        const int j = kb + e;
        const float sv = srow[j];
        const bool valid = (sv > 0.f) || (j == ibase + lr);  // sup2 = sup + I
        const float av = fast_tanh(s[j] + trow);
        p[run * 8 + e] = valid ? av : -100.0f;
      }
    }
    // online softmax: row stats live in lanes (lr) and (lr+16); combine halves
    float cm = p[0];
#pragma unroll
    for (int e = 1; e < 16; ++e) cm = fmaxf(cm, p[e]);
    cm = fmaxf(cm, __shfl_xor(cm, 16, 32));
    const float mnew  = fmaxf(mrun, cm);
    const float scale = __expf(mrun - mnew);
    float csum = 0.f;
    v16bf A;
#pragma unroll
    for (int e = 0; e < 16; ++e) {
      const float pe = __expf(p[e] - mnew);
      csum += pe;
      A[e] = (bf16_t)pe;
    }
    csum += __shfl_xor(csum, 16, 32);
    lrun = lrun * scale + csum;
    // rescale accumulators only when some row's max actually moved
    if (__ballot(mnew > mrun) != 0ull) {
#pragma unroll
      for (int m = 0; m < 8; ++m) {
        const float sc = __shfl(scale, m + hi * 8, 32);
        acc[0][m] *= sc; acc[1][m] *= sc; acc[2][m] *= sc; acc[3][m] *= sc;
      }
    }
    mrun = mnew;
    // B fragments from hT (col-major h, contiguous K runs of 8 bf16 = 16B)
#pragma unroll
    for (int nt = 0; nt < 4; ++nt) {
      const bf16_t* hp = hT + (size_t)(nt * 16 + lr) * NN + jb + kb0;
      const v8bf b0 = *reinterpret_cast<const v8bf*>(hp);
      const v8bf b1 = *reinterpret_cast<const v8bf*>(hp + 16);
      v16bf B;
#pragma unroll
      for (int e = 0; e < 8; ++e) { B[e] = b0[e]; B[8 + e] = b1[e]; }
      acc[nt] = wmma_bf16(A, B, acc[nt]);
    }
  }
  // epilogue: gat = tanh(acc / l); store transposed bf16 for the next GEMM
#pragma unroll
  for (int m = 0; m < 8; ++m) {
    const float linv = 1.0f / __shfl(lrun, m + hi * 8, 32);
    const int row = ibase + hi * 8 + m;
#pragma unroll
    for (int nt = 0; nt < 4; ++nt) {
      gatT[(size_t)(nt * 16 + lr) * NN + row] = (bf16_t)fast_tanh(acc[nt][m] * linv);
    }
  }
}

// ---------------------------------------------------------------------------
// K3: agg = (sup + I) @ gat_feat     (8192x8192x64), A built bf16 on the fly
// ---------------------------------------------------------------------------
__global__ void k_agg(const float* __restrict__ sup, const bf16_t* __restrict__ gatT,
                      float* __restrict__ agg) {
  const int lane  = threadIdx.x & 31;
  const int wv    = threadIdx.x >> 5;
  const int itile = blockIdx.x * 2 + wv;
  const int ibase = itile * 16;
  const int hi = lane >> 4, lr = lane & 15;
  const int kb0 = hi ? 8 : 0;

  const float* srow = sup + (size_t)(ibase + lr) * NN;
  v8f acc[4] = {};

  for (int jb = 0; jb < NN; jb += 32) {
    __builtin_prefetch(srow + jb + 512, 0, 1);
    v16bf A;
#pragma unroll
    for (int run = 0; run < 2; ++run) {
      const int kb = jb + kb0 + run * 16;
#pragma unroll
      for (int e = 0; e < 8; ++e) {
        const int j = kb + e;
        const float v = srow[j] + ((j == ibase + lr) ? 1.0f : 0.0f);
        A[run * 8 + e] = (bf16_t)v;
      }
    }
#pragma unroll
    for (int nt = 0; nt < 4; ++nt) {
      const bf16_t* gp = gatT + (size_t)(nt * 16 + lr) * NN + jb + kb0;
      const v8bf b0 = *reinterpret_cast<const v8bf*>(gp);
      const v8bf b1 = *reinterpret_cast<const v8bf*>(gp + 16);
      v16bf B;
#pragma unroll
      for (int e = 0; e < 8; ++e) { B[e] = b0[e]; B[8 + e] = b1[e]; }
      acc[nt] = wmma_bf16(A, B, acc[nt]);
    }
  }
#pragma unroll
  for (int m = 0; m < 8; ++m) {
    const int row = ibase + hi * 8 + m;
#pragma unroll
    for (int nt = 0; nt < 4; ++nt) {
      agg[(size_t)row * DGAT + nt * 16 + lr] = acc[nt][m];
    }
  }
}

// ---------------------------------------------------------------------------
// K4: out = normalize(relu(agg @ W_gcn)) ; deterministic row-norm reduction
// (shfl-xor tree inside each 16-lane half + fixed-order 4-wave LDS combine)
// ---------------------------------------------------------------------------
__global__ void k_out(const float* __restrict__ agg, const float* __restrict__ Wgcn,
                      float* __restrict__ out) {
  __shared__ float part[4][16];
  const int lane  = threadIdx.x & 31;
  const int wv    = threadIdx.x >> 5;          // 4 waves, one 16-col tile each
  const int ibase = blockIdx.x * 16;
  const int cbase = wv * 16;
  const int hi = lane >> 4, lr = lane & 15;
  const int kb0 = hi ? 8 : 0;

  const float* ar = agg + (size_t)(ibase + lr) * DGAT;
  v8f acc = {};
#pragma unroll
  for (int ch = 0; ch < 2; ++ch) {
    const int kb = ch * 32 + kb0;
    v16bf a, b;
#pragma unroll
    for (int e = 0; e < 8; ++e) {
      a[e]     = (bf16_t)ar[kb + e];
      a[8 + e] = (bf16_t)ar[kb + 16 + e];
      b[e]     = (bf16_t)Wgcn[(size_t)(kb + e) * DGCN + cbase + lr];
      b[8 + e] = (bf16_t)Wgcn[(size_t)(kb + 16 + e) * DGCN + cbase + lr];
    }
    acc = wmma_bf16(a, b, acc);
  }
#pragma unroll
  for (int m = 0; m < 8; ++m) acc[m] = fmaxf(acc[m], 0.f);

  // deterministic sum of squares over this wave's 16 columns
  float sq[8];
#pragma unroll
  for (int m = 0; m < 8; ++m) sq[m] = acc[m] * acc[m];
#pragma unroll
  for (int ofs = 1; ofs < 16; ofs <<= 1) {
#pragma unroll
    for (int m = 0; m < 8; ++m) sq[m] += __shfl_xor(sq[m], ofs, 32);
  }
  if (lane == 0) {
#pragma unroll
    for (int m = 0; m < 8; ++m) part[wv][m] = sq[m];        // rows 0..7
  }
  if (lane == 16) {
#pragma unroll
    for (int m = 0; m < 8; ++m) part[wv][8 + m] = sq[m];    // rows 8..15
  }
  __syncthreads();
#pragma unroll
  for (int m = 0; m < 8; ++m) {
    const int r = hi * 8 + m;
    const float tot = ((part[0][r] + part[1][r]) + (part[2][r] + part[3][r]));
    const float inv = 1.0f / fmaxf(sqrtf(tot), 1e-12f);
    out[(size_t)(ibase + r) * DGCN + cbase + lr] = acc[m] * inv;
  }
}

// ---------------------------------------------------------------------------
extern "C" void kernel_launch(void* const* d_in, const int* in_sizes, int n_in,
                              void* d_out, int out_size, void* d_ws, size_t ws_size,
                              hipStream_t stream) {
  (void)in_sizes; (void)n_in; (void)out_size; (void)ws_size;
  const float* feat = (const float*)d_in[0];   // [8192,128]
  const float* sup  = (const float*)d_in[1];   // [8192,8192]
  const float* Wmap = (const float*)d_in[2];   // [128,64]
  const float* bmap = (const float*)d_in[3];   // [1]
  const float* U    = (const float*)d_in[4];   // [64,1]
  const float* V    = (const float*)d_in[5];   // [64,1]
  const float* Wgcn = (const float*)d_in[6];   // [64,64]
  float* out = (float*)d_out;                  // [8192,64]

  char* ws = (char*)d_ws;                      // ~6.1 MB used
  float*  h    = (float*) (ws);                          // 2 MB
  bf16_t* hT   = (bf16_t*)(ws + (2u << 20));             // 1 MB
  bf16_t* gT   = (bf16_t*)(ws + (3u << 20));             // 1 MB
  float*  agg  = (float*) (ws + (4u << 20));             // 2 MB
  float*  s    = (float*) (ws + (6u << 20));             // 32 KB
  float*  t    = (float*) (ws + (6u << 20) + NN * 4u);   // 32 KB

  // 2048 tiles, 2 waves/block
  k_map   <<<1024, 64, 0, stream>>>(feat, Wmap, h, hT);
  k_scores<<<NN / 256, 256, 0, stream>>>(h, U, V, bmap, s, t);
  // 512 row-tiles, 2 waves/block
  k_att   <<<256, 64, 0, stream>>>(sup, s, t, hT, gT);
  k_agg   <<<256, 64, 0, stream>>>(sup, gT, agg);
  k_out   <<<512, 128, 0, stream>>>(agg, Wgcn, out);
}
